// PA_MOE_gate_40415642255843
// MI455X (gfx1250) — matrix operand order repaired and checked by reading, not compiled, not run on hardware
//
#include <hip/hip_runtime.h>
#include <hip/hip_bf16.h>

typedef __attribute__((ext_vector_type(16))) __bf16 v16bf;
typedef __attribute__((ext_vector_type(8)))  float  v8f;
typedef __attribute__((ext_vector_type(4)))  unsigned int u32x4;

struct FragBits { u32x4 lo, hi; };   // 32 bytes == v16bf

__device__ __forceinline__ v16bf make_frag(u32x4 lo, u32x4 hi) {
    FragBits fb{lo, hi};
    return __builtin_bit_cast(v16bf, fb);
}

// native bf16 convert (backend picks v_cvt_*_bf16 on gfx1250)
__device__ __forceinline__ unsigned short f2bf(float f) {
    __bf16 h = (__bf16)f;
    return __builtin_bit_cast(unsigned short, h);
}

#define ZERO8 {0.f,0.f,0.f,0.f,0.f,0.f,0.f,0.f}

// ---------------------------------------------------------------------------
// 1) Convert expert conv weights fp32 (O,C,k,k) -> bf16 per-tap [dy][dx][o][c]
// ---------------------------------------------------------------------------
__global__ __launch_bounds__(256)
void prep_weights(const float* __restrict__ ew0, const float* __restrict__ ew1,
                  const float* __restrict__ ew2, const float* __restrict__ ew3,
                  unsigned short* __restrict__ wb) {
    int gid = blockIdx.x * 256 + threadIdx.x;
    const int off1 = 4096, off2 = 40960, off3 = 241664, off4 = 737280;
    if (gid >= off4) return;
    const float* src; int k, rel;
    if      (gid < off1) { src = ew0; k = 1;  rel = gid;        }
    else if (gid < off2) { src = ew1; k = 3;  rel = gid - off1; }
    else if (gid < off3) { src = ew2; k = 7;  rel = gid - off2; }
    else                 { src = ew3; k = 11; rel = gid - off3; }
    int tap = rel >> 12, oc = rel & 4095;
    int o = oc >> 6, c = oc & 63;
    int dy = tap / k, dx = tap % k;
    wb[gid] = f2bf(src[((o * 64 + c) * k + dy) * k + dx]);
}

// ---------------------------------------------------------------------------
// 2) Fused gating: h1=relu(W1@x), h2=relu(W2@h1), 16x16 avg-pool -> pooled
//    One block per pooled cell (b,py,px): 256 pixels, 8 waves.
// ---------------------------------------------------------------------------
__global__ __launch_bounds__(256)
void gating_fused(const float* __restrict__ x, const float* __restrict__ w1,
                  const float* __restrict__ b1, const float* __restrict__ w2,
                  const float* __restrict__ b2, float* __restrict__ pooled) {
    const int px = blockIdx.x, py = blockIdx.y, b = blockIdx.z;
    const int tid = threadIdx.x;

    __shared__ __align__(16) unsigned short xs [256 * 64];  // [pixel][c]
    __shared__ __align__(16) unsigned short w1s[512 * 64];  // [o][c]
    __shared__ __align__(16) unsigned short w2s[16 * 512];  // [o][m]
    __shared__ __align__(16) unsigned short h1s[256 * 64];  // [pixel][m_local]
    __shared__ float red[16];
    if (tid < 16) red[tid] = 0.f;

    for (int idx = tid; idx < 256 * 64; idx += 256) {
        int p = idx & 255, c = idx >> 8;
        int Y = py * 16 + (p >> 4), X = px * 16 + (p & 15);
        xs[p * 64 + c] = f2bf(x[(((long)b * 64 + c) * 256 + Y) * 256 + X]);
    }
    for (int idx = tid; idx < 512 * 64; idx += 256) w1s[idx] = f2bf(w1[idx]);
    for (int idx = tid; idx < 16 * 512; idx += 256) w2s[idx] = f2bf(w2[idx]);
    __syncthreads();

    const int w = tid >> 5, lane = tid & 31;
    const int lg = lane >> 4, lr = lane & 15;
    v8f accH0 = ZERO8, accH1 = ZERO8;     // h2 accumulators: N-cols [w*32, w*32+32)

    for (int ch = 0; ch < 8; ++ch) {      // 64-row chunks of h1
        // ---- GEMM1: h1_chunk = W1[ch] @ x  (M=64,N=256,K=64) ----
        const int mt = w >> 1;
        for (int j = 0; j < 8; ++j) {
            const int nt = (w & 1) * 8 + j;
            v8f acc = ZERO8;
#pragma unroll
            for (int kb = 0; kb < 2; ++kb) {
                int abase = (ch * 64 + mt * 16 + lr) * 64 + kb * 32 + lg * 16;
                u32x4 alo = *(const u32x4*)(w1s + abase);
                u32x4 ahi = *(const u32x4*)(w1s + abase + 8);
                int bbase = (nt * 16 + lr) * 64 + kb * 32 + lg * 16;
                u32x4 blo = *(const u32x4*)(xs + bbase);
                u32x4 bhi = *(const u32x4*)(xs + bbase + 8);
                acc = __builtin_amdgcn_wmma_f32_16x16x32_bf16(
                    false, make_frag(alo, ahi), false, make_frag(blo, bhi),
                    (short)0, acc, false, false);
            }
#pragma unroll
            for (int r = 0; r < 8; ++r) {   // bias + relu -> bf16 LDS
                int m = mt * 16 + r + lg * 8;
                float v = acc[r] + b1[ch * 64 + m];
                v = v > 0.f ? v : 0.f;
                h1s[(nt * 16 + lr) * 64 + m] = f2bf(v);
            }
        }
        __syncthreads();
        // ---- GEMM2 partial: h2 += W2[:,chunk] @ h1_chunk ----
#pragma unroll
        for (int t2 = 0; t2 < 2; ++t2) {
            v8f* accp = t2 ? &accH1 : &accH0;
#pragma unroll
            for (int kb = 0; kb < 2; ++kb) {
                int abase = lr * 512 + ch * 64 + kb * 32 + lg * 16;
                u32x4 alo = *(const u32x4*)(w2s + abase);
                u32x4 ahi = *(const u32x4*)(w2s + abase + 8);
                int bbase = (w * 32 + t2 * 16 + lr) * 64 + kb * 32 + lg * 16;
                u32x4 blo = *(const u32x4*)(h1s + bbase);
                u32x4 bhi = *(const u32x4*)(h1s + bbase + 8);
                *accp = __builtin_amdgcn_wmma_f32_16x16x32_bf16(
                    false, make_frag(alo, ahi), false, make_frag(blo, bhi),
                    (short)0, *accp, false, false);
            }
        }
        __syncthreads();
    }
    // ---- bias + relu + pool reduce ----
#pragma unroll
    for (int t2 = 0; t2 < 2; ++t2) {
        v8f a = t2 ? accH1 : accH0;
#pragma unroll
        for (int r = 0; r < 8; ++r) {
            int m = r + lg * 8;
            float v = a[r] + b2[m];
            v = v > 0.f ? v : 0.f;
            atomicAdd(&red[m], v);
        }
    }
    __syncthreads();
    if (tid < 16)
        pooled[((b * 16 + tid) * 16 + py) * 16 + px] = red[tid] * (1.f / 256.f);
}

// ---------------------------------------------------------------------------
// 3) Gating tail: w3 conv, logits, top-2 softmax gates, cv^2 loss. One block.
// ---------------------------------------------------------------------------
__global__ __launch_bounds__(256)
void gating_finalize(const float* __restrict__ pooled, const float* __restrict__ w3,
                     const float* __restrict__ b3, const float* __restrict__ w_gate,
                     float* __restrict__ feats, float* __restrict__ gates,
                     float* __restrict__ loss_out) {
    int tid = threadIdx.x;
    for (int idx = tid; idx < 8 * 16 * 256; idx += 256) {
        int b = idx >> 12, c = (idx >> 8) & 15, p = idx & 255;
        float s = b3[c];
        for (int cc = 0; cc < 16; ++cc)
            s += w3[c * 16 + cc] * pooled[((b * 16 + cc) << 8) + p];
        feats[b * 4096 + c * 256 + p] = s;
    }
    __threadfence();
    __syncthreads();
    __shared__ float logits[32];
    if (tid < 32) {
        int b = tid >> 2, e = tid & 3;
        float s = 0.f;
        for (int f = 0; f < 4096; ++f) s += feats[b * 4096 + f] * w_gate[f * 4 + e];
        logits[tid] = s;
    }
    __syncthreads();
    if (tid == 0) {
        float g[8][4];
        for (int b = 0; b < 8; ++b)
            for (int e = 0; e < 4; ++e) g[b][e] = 0.f;
        for (int b = 0; b < 8; ++b) {
            const float* L = logits + b * 4;
            int i1 = 0;
            for (int e = 1; e < 4; ++e) if (L[e] > L[i1]) i1 = e;
            int i2 = (i1 == 0) ? 1 : 0;
            for (int e = 0; e < 4; ++e)
                if (e != i1 && L[e] > L[i2]) i2 = e;
            float e2 = __expf(L[i2] - L[i1]);
            float d = 1.f + e2;
            g[b][i1] = 1.f / d;
            g[b][i2] = e2 / d;
        }
        float imp[4] = {0, 0, 0, 0}, ld[4] = {0, 0, 0, 0};
        for (int e = 0; e < 4; ++e)
            for (int b = 0; b < 8; ++b) {
                imp[e] += g[b][e];
                ld[e] += (g[b][e] > 0.f) ? 1.f : 0.f;
            }
        auto cv2 = [](const float* v) {
            float mu = (v[0] + v[1] + v[2] + v[3]) * 0.25f;
            float s = 0.f;
            for (int i = 0; i < 4; ++i) { float d = v[i] - mu; s += d * d; }
            float var = s / 3.f;                    // ddof=1
            return var / (mu * mu + 1e-10f);
        };
        loss_out[0] = (cv2(imp) + cv2(ld)) * 0.1f;
        for (int b = 0; b < 8; ++b)
            for (int e = 0; e < 4; ++e) gates[b * 4 + e] = g[b][e];
    }
}

// ---------------------------------------------------------------------------
// 4) Expert conv (implicit GEMM, per-tap 64x64 WMMA) + sigmoid gate fuse.
//    Block: 64 out-channels x 64 pixels. 8 waves; each wave = 1 M-tile x 2
//    N-tiles -> two independent WMMA chains, A fragment reused 2x.
// ---------------------------------------------------------------------------
template <int KS, bool FIRST>
__global__ __launch_bounds__(256)
void expert_conv(const float* __restrict__ x, const unsigned short* __restrict__ wtap,
                 const float* __restrict__ bias, const float* __restrict__ gates,
                 int e, float* __restrict__ out) {
    constexpr int PAD = KS / 2;
    constexpr int NPX = 64;
    constexpr int WT = NPX + KS - 1;
    const int x0 = blockIdx.x * NPX, y = blockIdx.y, b = blockIdx.z;
    const int tid = threadIdx.x;
    const float g = gates[b * 4 + e];
    if (!FIRST && g == 0.f) return;                       // top-2 sparsity skip
    if (FIRST && g == 0.f) {                              // still init out tile
        for (int idx = tid; idx < 64 * NPX; idx += 256) {
            int m = idx >> 6, n = idx & (NPX - 1);
            out[(((long)b * 64 + m) * 256 + y) * 256 + x0 + n] = 0.f;
        }
        return;
    }

    __shared__ __align__(16) unsigned short tile[KS * WT * 64];  // [row][col][c]
    for (int idx = tid; idx < 64 * KS * WT; idx += 256) {
        int col = idx % WT, t = idx / WT;
        int row = t % KS, c = t / KS;
        int gy = y + row - PAD, gx = x0 + col - PAD;
        float v = 0.f;
        if (gy >= 0 && gy < 256 && gx >= 0 && gx < 256)
            v = x[(((long)b * 64 + c) * 256 + gy) * 256 + gx];
        tile[(row * WT + col) * 64 + c] = f2bf(v);
    }
    __syncthreads();

    const int w = tid >> 5, lane = tid & 31;
    const int lg = lane >> 4, lr = lane & 15;
    const int m0 = (w >> 1) * 16;
    const int n0 = (w & 1) * 32;          // two N-tiles: n0, n0+16
    v8f acc0 = ZERO8, acc1 = ZERO8;

    for (int dy = 0; dy < KS; ++dy)
        for (int dx = 0; dx < KS; ++dx) {
            const int tap = dy * KS + dx;
#pragma unroll
            for (int kb = 0; kb < 2; ++kb) {
                int abase = (tap * 64 + m0 + lr) * 64 + kb * 32 + lg * 16;
                u32x4 alo = *(const u32x4*)(wtap + abase);
                u32x4 ahi = *(const u32x4*)(wtap + abase + 8);
                v16bf A = make_frag(alo, ahi);
                int bb0 = (dy * WT + n0 + lr + dx) * 64 + kb * 32 + lg * 16;
                u32x4 b0lo = *(const u32x4*)(tile + bb0);
                u32x4 b0hi = *(const u32x4*)(tile + bb0 + 8);
                int bb1 = bb0 + 16 * 64;
                u32x4 b1lo = *(const u32x4*)(tile + bb1);
                u32x4 b1hi = *(const u32x4*)(tile + bb1 + 8);
                acc0 = __builtin_amdgcn_wmma_f32_16x16x32_bf16(
                    false, A, false, make_frag(b0lo, b0hi), (short)0, acc0, false, false);
                acc1 = __builtin_amdgcn_wmma_f32_16x16x32_bf16(
                    false, A, false, make_frag(b1lo, b1hi), (short)0, acc1, false, false);
            }
        }

#pragma unroll
    for (int t2 = 0; t2 < 2; ++t2) {      // C layout: m = m0+r+8*lg, n = lane%16
        v8f a = t2 ? acc1 : acc0;
        int n = n0 + t2 * 16 + lr;
#pragma unroll
        for (int r = 0; r < 8; ++r) {
            int m = m0 + r + lg * 8;
            long oi = (((long)b * 64 + m) * 256 + y) * 256 + x0 + n;
            float v = a[r] + bias[m];
            float attn = 1.f / (1.f + __expf(-v));
            float contrib = g * attn * x[oi];
            if (FIRST) out[oi] = contrib;
            else       out[oi] += contrib;
        }
    }
}

// ---------------------------------------------------------------------------
extern "C" void kernel_launch(void* const* d_in, const int* in_sizes, int n_in,
                              void* d_out, int out_size, void* d_ws, size_t ws_size,
                              hipStream_t stream) {
    const float* x      = (const float*)d_in[0];
    const float* w1     = (const float*)d_in[1];
    const float* b1     = (const float*)d_in[2];
    const float* w2     = (const float*)d_in[3];
    const float* b2     = (const float*)d_in[4];
    const float* w3     = (const float*)d_in[5];
    const float* b3     = (const float*)d_in[6];
    const float* w_gate = (const float*)d_in[7];
    const float* ew0 = (const float*)d_in[8];
    const float* eb0 = (const float*)d_in[9];
    const float* ew1 = (const float*)d_in[10];
    const float* eb1 = (const float*)d_in[11];
    const float* ew2 = (const float*)d_in[12];
    const float* eb2 = (const float*)d_in[13];
    const float* ew3 = (const float*)d_in[14];
    const float* eb3 = (const float*)d_in[15];
    float* out = (float*)d_out;

    char* ws = (char*)d_ws;
    float* pooled = (float*)(ws + 0);              // 32768 f32
    float* feats  = (float*)(ws + 131072);         // 32768 f32
    float* gates  = (float*)(ws + 262144);         // 32 f32
    unsigned short* wb = (unsigned short*)(ws + 262272);  // 737280 bf16

    prep_weights<<<2880, 256, 0, stream>>>(ew0, ew1, ew2, ew3, wb);
    gating_fused<<<dim3(16, 16, 8), 256, 0, stream>>>(x, w1, b1, w2, b2, pooled);
    gating_finalize<<<1, 256, 0, stream>>>(pooled, w3, b3, w_gate, feats, gates,
                                           out + (out_size - 1));
    dim3 eg(4, 256, 8);
    expert_conv<1,  true ><<<eg, 256, 0, stream>>>(x, wb + 0,      eb0, gates, 0, out);
    expert_conv<3,  false><<<eg, 256, 0, stream>>>(x, wb + 4096,   eb1, gates, 1, out);
    expert_conv<7,  false><<<eg, 256, 0, stream>>>(x, wb + 40960,  eb2, gates, 2, out);
    expert_conv<11, false><<<eg, 256, 0, stream>>>(x, wb + 241664, eb3, gates, 3, out);
}
